// GRU_D_53987738911252
// MI455X (gfx1250) — matrix operand order repaired and checked by reading, not compiled, not run on hardware
//
#include <hip/hip_runtime.h>
#include <hip/hip_bf16.h>

typedef __attribute__((ext_vector_type(16))) _Float16 v16h;
typedef __attribute__((ext_vector_type(8)))  _Float16 v8h;
typedef __attribute__((ext_vector_type(8)))  float    v8f;

#define B_TOT 16384
#define D_IN  32
#define H_DIM 64
#define T_OBS 100
#define N_EX  20
#define BM    32      // batch rows per workgroup
#define NT    256     // 8 wave32s

// ---- fragment loaders -------------------------------------------------------
// A fragment (16x32 f16 tile) from LDS, row-major with stride 128 halves.
// CDNA5 16-bit A layout: lanes 0-15 hold K {0..7, 16..23}; lanes 16-31 hold
// K {8..15, 24..31} (two contiguous 16B chunks per lane -> ds_load_b128 x2).
__device__ __forceinline__ v16h lds_load_A(const _Float16* base, int mt, int kbase, int lane) {
  int row = mt * 16 + (lane & 15);
  int sel = (lane & 16) ? 8 : 0;
  const v8h* p0 = (const v8h*)(base + row * 128 + kbase + sel);
  const v8h* p1 = (const v8h*)(base + row * 128 + kbase + 16 + sel);
  v8h lo = *p0, hi = *p1;
  v16h a;
#pragma unroll
  for (int i = 0; i < 8; ++i) { a[i] = lo[i]; a[i + 8] = hi[i]; }
  return a;
}

// B fragment (32x16 f16 tile) from global f32 weights W[k*N + n] (fan_in x fan_out).
// Lane holds column n = nt*16 + (lane&15); lanes 0-15: K kbase+{0..7,16..23},
// lanes 16-31: K kbase+{8..15,24..31}. Loaded ONCE, lives in VGPRs for all 120 steps.
__device__ __forceinline__ v16h load_B_global(const float* W, int N, int nt, int kbase, int lane) {
  int col = nt * 16 + (lane & 15);
  int k0  = kbase + ((lane & 16) ? 8 : 0);
  v16h b;
#pragma unroll
  for (int i = 0; i < 16; ++i) b[i] = (_Float16)W[(size_t)(k0 + i + (i >= 8 ? 8 : 0)) * N + col];
  return b;
}

__device__ __forceinline__ v8f splat8(float x) {
  v8f c;
#pragma unroll
  for (int i = 0; i < 8; ++i) c[i] = x;
  return c;
}

// Fast gate activations: single v_exp_f32 + single v_rcp_f32 each
// (avoids the IEEE div_scale/div_fmas/div_fixup expansion seen in round 1).
__device__ __forceinline__ float sigmoid_f(float x) {
  return __builtin_amdgcn_rcpf(1.f + __expf(-x));
}
__device__ __forceinline__ float tanh_f(float x) {
  // tanh(x) = 1 - 2/(exp(2x)+1)
  return __builtin_fmaf(-2.f, __builtin_amdgcn_rcpf(__expf(2.f * x) + 1.f), 1.f);
}

// ---- persistent GRU-D kernel ------------------------------------------------
__global__ __launch_bounds__(NT) void grud_persistent(
    const float* __restrict__ obs_times, const float* __restrict__ obs_mask,
    const float* __restrict__ Mg,        const float* __restrict__ Xg,
    const float* __restrict__ w_gx,      const float* __restrict__ b_gx,
    const float* __restrict__ w_gh,      const float* __restrict__ b_gh,
    const float* __restrict__ Wr,        const float* __restrict__ br,
    const float* __restrict__ Wz,        const float* __restrict__ bz,
    const float* __restrict__ Wh,        const float* __restrict__ bh,
    const float* __restrict__ W1,        const float* __restrict__ b1,
    const float* __restrict__ W2,        const float* __restrict__ b2,
    float* __restrict__ out, float* __restrict__ loss_part)
{
  __shared__ __attribute__((aligned(16))) _Float16 s_inp[BM * 128]; // A staging (f16)
  __shared__ float s_rz[BM * 128];     // r|z, later h_tilde in cols 0..63
  __shared__ float s_h[BM * H_DIM];
  __shared__ float s_hhat[BM * H_DIM];
  __shared__ float s_p[BM * D_IN];
  __shared__ float s_lastx[BM * D_IN];
  __shared__ float s_X[BM * D_IN];
  __shared__ float s_Mv[BM * D_IN];
  __shared__ float s_obs[BM];
  __shared__ float s_lastt[BM];
  __shared__ float s_gx[BM];
  __shared__ float s_gh[BM];
  __shared__ float s_red[NT];

  const int tid  = threadIdx.x;
  const int lane = tid & 31;
  const int wv   = tid >> 5;       // wave id 0..7
  const int b0   = blockIdx.x * BM;

  // ---- load weight B-fragments into registers (reused every step) ----
  v16h brz[4], bhf[4], b1f[2], b2f[2];
  {
    const float* Wrz = (wv < 4) ? Wr : Wz;   // waves 0-3 -> r, 4-7 -> z
#pragma unroll
    for (int k = 0; k < 4; ++k) brz[k] = load_B_global(Wrz, H_DIM, wv & 3, k * 32, lane);
#pragma unroll
    for (int k = 0; k < 4; ++k) bhf[k] = load_B_global(Wh, H_DIM, wv >> 1, k * 32, lane);
#pragma unroll
    for (int k = 0; k < 2; ++k) b1f[k] = load_B_global(W1, H_DIM, wv >> 1, k * 32, lane);
#pragma unroll
    for (int k = 0; k < 2; ++k) {
      if (wv < 4) b2f[k] = load_B_global(W2, D_IN, wv >> 1, k * 32, lane);
      else {
        v16h z;
#pragma unroll
        for (int i = 0; i < 16; ++i) z[i] = (_Float16)0.f;
        b2f[k] = z;
      }
    }
  }

  const float wgx = w_gx[0], bgx0 = b_gx[0];
  const float wgh = w_gh[0], bgh0 = b_gh[0];
  const float t_last_obs = obs_times[T_OBS - 1];

  for (int i = tid; i < BM * H_DIM; i += NT) s_h[i] = 0.f;
  for (int i = tid; i < BM * D_IN; i += NT) s_lastx[i] = 0.f;
  if (tid < BM) s_lastt[tid] = 0.f;
  float loss = 0.f;
  __syncthreads();

  for (int t = 0; t < T_OBS + N_EX; ++t) {
    const bool  op   = (t < T_OBS);
    const float tcur = op ? obs_times[t] : (t_last_obs + 0.1f * (float)(t - T_OBS + 1));

    // ---- P0: per-row scalars + stream X/M tile; stage h (f16) for p-model ----
    if (tid < BM) {
      int r = tid;
      float ob = op ? obs_mask[(size_t)t * B_TOT + b0 + r] : 0.f;
      s_obs[r] = ob;
      float iv = tcur - s_lastt[r];
      s_gx[r] = __expf(-fmaxf(iv * wgx + bgx0, 0.f));
      s_gh[r] = __expf(-fmaxf(iv * wgh + bgh0, 0.f));
      if (ob > 0.f) s_lastt[r] = tcur;
    }
    if (op) {
#pragma unroll
      for (int i = 0; i < 4; ++i) {
        int idx = tid + i * NT;               // 32x32
        int r = idx >> 5, d = idx & 31;
        size_t g = (size_t)t * B_TOT * D_IN + (size_t)(b0 + r) * D_IN + d;
        s_X[idx]  = Xg[g];
        s_Mv[idx] = Mg[g];
      }
#pragma unroll
      for (int i = 0; i < 8; ++i) {           // h_{t-1} -> f16 (A for W1 GEMM)
        int idx = tid + i * NT;               // 32x64
        int r = idx >> 6, c = idx & 63;
        s_inp[r * 128 + c] = (_Float16)s_h[idx];
      }
    }
    __syncthreads();

    // ---- p = relu(h @ W1 + b1) @ W2 + b2  (h = h_{t-1}) ----
    if (op) {
      {
        int mt = wv & 1, nt = wv >> 1;        // 2x4 tiles, 1 per wave
        v8f c = splat8(b1[nt * 16 + (lane & 15)]);
#pragma unroll
        for (int k = 0; k < 2; ++k) {
          v16h a = lds_load_A(s_inp, mt, k * 32, lane);
          c = __builtin_amdgcn_wmma_f32_16x16x32_f16(false, a, false, b1f[k], (short)0, c, false, false);
        }
#pragma unroll
        for (int i = 0; i < 8; ++i) {
          int row = mt * 16 + i + ((lane >> 4) << 3);
          int col = nt * 16 + (lane & 15);
          s_inp[row * 128 + 64 + col] = (_Float16)fmaxf(c[i], 0.f);  // a1 -> cols 64..127
        }
      }
      __syncthreads();
      if (wv < 4) {                            // 2x2 tiles, waves 0..3 (wave-uniform)
        int mt = wv & 1, nt = wv >> 1;
        v8f c = splat8(b2[nt * 16 + (lane & 15)]);
#pragma unroll
        for (int k = 0; k < 2; ++k) {
          v16h a = lds_load_A(s_inp + 64, mt, k * 32, lane);
          c = __builtin_amdgcn_wmma_f32_16x16x32_f16(false, a, false, b2f[k], (short)0, c, false, false);
        }
#pragma unroll
        for (int i = 0; i < 8; ++i) {
          int row = mt * 16 + i + ((lane >> 4) << 3);
          int col = nt * 16 + (lane & 15);
          s_p[row * D_IN + col] = c[i];
        }
      }
      __syncthreads();
    }

    // ---- elementwise: loss, last_x, X_hat, M_obs, h_hat; build inp ----
#pragma unroll
    for (int i = 0; i < 4; ++i) {
      int idx = tid + i * NT;                  // 32x32
      int r = idx >> 5, d = idx & 31;
      float ob = s_obs[r];
      float m  = s_Mv[idx];
      float mo = ob * m;                       // 0 in propagation phase
      float lx = s_lastx[idx];
      if (op) {
        float x = s_X[idx];
        loss += mo * fabsf(x - s_p[idx]);
        if (ob > 0.f) { lx = lx * (1.f - m) + x; s_lastx[idx] = lx; }
      }
      float xh = mo * lx + (1.f - mo) * (s_gx[r] * lx);
      s_inp[r * 128 + d]      = (_Float16)xh;  // cols 0..31
      s_inp[r * 128 + 96 + d] = (_Float16)mo;  // cols 96..127
    }
#pragma unroll
    for (int i = 0; i < 8; ++i) {
      int idx = tid + i * NT;                  // 32x64
      int r = idx >> 6, c = idx & 63;
      float hh = s_gh[r] * s_h[idx];
      s_hhat[idx] = hh;
      s_inp[r * 128 + 32 + c] = (_Float16)hh;  // cols 32..95
    }
    __syncthreads();

    // ---- r|z = sigmoid(inp @ [Wr|Wz] + [br|bz]) : 16 tiles, 2 per wave ----
    {
      const float* bb = (wv < 4) ? br : bz;
      float bias = bb[(wv & 3) * 16 + (lane & 15)];
#pragma unroll
      for (int mt = 0; mt < 2; ++mt) {
        v8f c = splat8(bias);
#pragma unroll
        for (int k = 0; k < 4; ++k) {
          v16h a = lds_load_A(s_inp, mt, k * 32, lane);
          c = __builtin_amdgcn_wmma_f32_16x16x32_f16(false, a, false, brz[k], (short)0, c, false, false);
        }
#pragma unroll
        for (int i = 0; i < 8; ++i) {
          int row = mt * 16 + i + ((lane >> 4) << 3);
          int col = wv * 16 + (lane & 15);     // waves 0-3 -> r (cols 0..63), 4-7 -> z
          s_rz[row * 128 + col] = sigmoid_f(c[i]);
        }
      }
    }
    __syncthreads();

    // ---- inp2: h_hat slice -> r * h_hat ----
#pragma unroll
    for (int i = 0; i < 8; ++i) {
      int idx = tid + i * NT;
      int r = idx >> 6, c = idx & 63;
      s_inp[r * 128 + 32 + c] = (_Float16)(s_rz[r * 128 + c] * s_hhat[idx]);
    }
    __syncthreads();

    // ---- h_tilde = tanh(inp2 @ Wh + bh) -> s_rz cols 0..63 ----
    {
      int mt = wv & 1, nt = wv >> 1;
      v8f c = splat8(bh[nt * 16 + (lane & 15)]);
#pragma unroll
      for (int k = 0; k < 4; ++k) {
        v16h a = lds_load_A(s_inp, mt, k * 32, lane);
        c = __builtin_amdgcn_wmma_f32_16x16x32_f16(false, a, false, bhf[k], (short)0, c, false, false);
      }
#pragma unroll
      for (int i = 0; i < 8; ++i) {
        int row = mt * 16 + i + ((lane >> 4) << 3);
        int col = nt * 16 + (lane & 15);
        s_rz[row * 128 + col] = tanh_f(c[i]);
      }
    }
    __syncthreads();

    // ---- h = (1-z)*h_hat + z*h_tilde ----
#pragma unroll
    for (int i = 0; i < 8; ++i) {
      int idx = tid + i * NT;
      int r = idx >> 6, c = idx & 63;
      float z  = s_rz[r * 128 + 64 + c];
      float ht = s_rz[r * 128 + c];
      s_h[idx] = (1.f - z) * s_hhat[idx] + z * ht;
    }
    __syncthreads();
  }

  // ---- outputs: h, and deterministic per-block loss partial ----
#pragma unroll
  for (int i = 0; i < 8; ++i) {
    int idx = tid + i * NT;
    int r = idx >> 6, c = idx & 63;
    out[1 + (size_t)(b0 + r) * H_DIM + c] = s_h[idx];
  }
  s_red[tid] = loss;
  __syncthreads();
  for (int s = NT / 2; s > 0; s >>= 1) {
    if (tid < s) s_red[tid] += s_red[tid + s];
    __syncthreads();
  }
  if (tid == 0) loss_part[blockIdx.x] = s_red[0];
}

// Deterministic final reduction of 512 block partials -> out[0]
__global__ void grud_reduce(const float* __restrict__ part, float* __restrict__ out, int n) {
  __shared__ float s[256];
  int tid = threadIdx.x;
  float v = 0.f;
  for (int i = tid; i < n; i += 256) v += part[i];
  s[tid] = v;
  __syncthreads();
  for (int k = 128; k > 0; k >>= 1) {
    if (tid < k) s[tid] += s[tid + k];
    __syncthreads();
  }
  if (tid == 0) out[0] = s[0];
}

extern "C" void kernel_launch(void* const* d_in, const int* in_sizes, int n_in,
                              void* d_out, int out_size, void* d_ws, size_t ws_size,
                              hipStream_t stream) {
  (void)in_sizes; (void)n_in; (void)out_size; (void)ws_size;
  const float* obs_times = (const float*)d_in[0];
  const float* obs_mask  = (const float*)d_in[1];
  const float* Mg        = (const float*)d_in[2];
  const float* Xg        = (const float*)d_in[3];
  const float* w_gx      = (const float*)d_in[4];
  const float* b_gx      = (const float*)d_in[5];
  const float* w_gh      = (const float*)d_in[6];
  const float* b_gh      = (const float*)d_in[7];
  const float* Wr        = (const float*)d_in[8];
  const float* br        = (const float*)d_in[9];
  const float* Wz        = (const float*)d_in[10];
  const float* bz        = (const float*)d_in[11];
  const float* Wh        = (const float*)d_in[12];
  const float* bh        = (const float*)d_in[13];
  const float* W1        = (const float*)d_in[14];
  const float* b1        = (const float*)d_in[15];
  const float* W2        = (const float*)d_in[16];
  const float* b2        = (const float*)d_in[17];

  float* out  = (float*)d_out;
  float* part = (float*)d_ws;      // 512 floats of scratch

  const int nblk = B_TOT / BM;     // 512 workgroups, 32 batch rows each
  grud_persistent<<<nblk, NT, 0, stream>>>(obs_times, obs_mask, Mg, Xg,
                                           w_gx, b_gx, w_gh, b_gh,
                                           Wr, br, Wz, bz, Wh, bh,
                                           W1, b1, W2, b2, out, part);
  grud_reduce<<<1, 256, 0, stream>>>(part, out, nblk);
}
